// BertTest_33483565040321
// MI455X (gfx1250) — matrix operand order repaired and checked by reading, not compiled, not run on hardware
//
#include <hip/hip_runtime.h>
#include <hip/hip_bf16.h>

#define SDIM   2048
#define BDIM   2
#define HDIM   1024
#define NHEAD  16
#define HD     64
#define MTOK   (SDIM * BDIM)   // 4096 tokens
#define NLAYER 12

typedef __attribute__((ext_vector_type(16))) _Float16 v16h;
typedef __attribute__((ext_vector_type(8)))  float    v8f;
typedef __attribute__((ext_vector_type(4)))  int      v4i_;

// ---------------------------------------------------------------------------
// Async global->LDS copy (CDNA5 GLOBAL_LOAD_ASYNC_TO_LDS_B128, ASYNCcnt).
// Falls back to a synchronous copy with identical barrier semantics.
// Builtin signature (from hipcc diagnostic): (int4 AS1*, int4 AS3*, Ii, Ii)
// ---------------------------------------------------------------------------
#if defined(__has_builtin)
#if __has_builtin(__builtin_amdgcn_global_load_async_to_lds_b128) && \
    __has_builtin(__builtin_amdgcn_s_wait_asynccnt)
#define ASYNC_LDS 1
#endif
#endif
#ifndef ASYNC_LDS
#define ASYNC_LDS 0
#endif

typedef __attribute__((address_space(1))) v4i_ gv4i;
typedef __attribute__((address_space(3))) v4i_ lv4i;

__device__ __forceinline__ void g2l_16B(_Float16* l, const _Float16* g) {
#if ASYNC_LDS
    __builtin_amdgcn_global_load_async_to_lds_b128(
        (gv4i*)(void*)g, (lv4i*)(void*)l, 0, 0);
#else
    *(uint4*)l = *(const uint4*)g;
#endif
}
__device__ __forceinline__ void wait_async() {
#if ASYNC_LDS
    __builtin_amdgcn_s_wait_asynccnt(0);
#endif
}

// ---------------------------------------------------------------------------
// WMMA fragment loaders (CDNA5 16-bit layouts, wave32)
// A 16x32 (MxK): lanes 0-15 -> M=lane, K 0..7 in v0-3, K 16..23 in v4-7
//                lanes16-31 -> M=lane-16, K 8..15 / 24..31
// ---------------------------------------------------------------------------
__device__ __forceinline__ v16h frag_a(const _Float16* base, int ld) {
    const int lane = threadIdx.x & 31;
    const int row  = lane & 15;
    const int kb   = (lane >> 4) << 3;           // 0 or 8
    const _Float16* p = base + row * ld + kb;
    v16h a;
#pragma unroll
    for (int j = 0; j < 8; ++j) a[j] = p[j];           // K = kb .. kb+7
#pragma unroll
    for (int j = 0; j < 8; ++j) a[8 + j] = p[16 + j];  // K = kb+16 .. kb+23
    return a;
}

// B 32x16 (KxN) sourced from a row-major (N,K) buffer: lane -> col = lane&15,
// lanes 0-15 hold K 0..7 / 16..23, lanes 16-31 hold K 8..15 / 24..31
__device__ __forceinline__ v16h frag_b(const _Float16* base, int ld) {
    const int lane = threadIdx.x & 31;
    const int col  = lane & 15;
    const int kb   = (lane >> 4) << 3;           // 0 or 8
    const _Float16* p = base + col * ld + kb;
    v16h b;
#pragma unroll
    for (int j = 0; j < 8; ++j) b[j] = p[j];
#pragma unroll
    for (int j = 0; j < 8; ++j) b[8 + j] = p[16 + j];
    return b;
}

__device__ __forceinline__ v8f wmma_f16(v16h a, v16h b, v8f c) {
    return __builtin_amdgcn_wmma_f32_16x16x32_f16(false, a, false, b,
                                                  (short)0, c, false, false);
}

// reductions within a 16-lane half of the wave32 (rows of a C tile)
__device__ __forceinline__ float halfmax(float v) {
#pragma unroll
    for (int m = 1; m <= 8; m <<= 1) v = fmaxf(v, __shfl_xor(v, m, 32));
    return v;
}
__device__ __forceinline__ float halfsum(float v) {
#pragma unroll
    for (int m = 1; m <= 8; m <<= 1) v += __shfl_xor(v, m, 32);
    return v;
}

// ---------------------------------------------------------------------------
// fp32 -> f16 convert (activations)
// ---------------------------------------------------------------------------
__global__ __launch_bounds__(256) void cvt_f32_f16(const float* __restrict__ x,
                                                   _Float16* __restrict__ y, int n) {
    int i = blockIdx.x * 256 + threadIdx.x;
    if (i < n) y[i] = (_Float16)x[i];
}

// W (K,N) fp32 -> Wt (N,K) f16, 32x32 LDS tile transpose
__global__ __launch_bounds__(256) void cvt_w_t(const float* __restrict__ w,
                                               _Float16* __restrict__ wt) {
    __shared__ _Float16 tile[32][33];
    const int bn = blockIdx.x * 32;   // N (out) tile
    const int bk = blockIdx.y * 32;   // K (in) tile
    const int tx = threadIdx.x & 31;
    const int ty = threadIdx.x >> 5;  // 8 rows / pass
#pragma unroll
    for (int r = ty; r < 32; r += 8)
        tile[r][tx] = (_Float16)w[(bk + r) * HDIM + bn + tx];
    __syncthreads();
#pragma unroll
    for (int r = ty; r < 32; r += 8)
        wt[(bn + r) * HDIM + bk + tx] = tile[tx][r];
}

// ---------------------------------------------------------------------------
// WMMA GEMM:  C(M x N=1024) = A16(M x 1024) * Wt(N x 1024)^T  (+bias, epilogue)
// block = 256 thr (8 waves as 4M x 2N), tile 128(M) x 128(N), K-chunks of 32.
// Wave tile 32x64 -> 8 WMMA per chunk, all fragments preloaded so the WMMAs
// issue back-to-back behind a single DS wait. Double-buffered async staging.
// MODE 0: f16 store (B,NH,S,HD)   (Q, K)
// MODE 1: f16 store (B,NH,HD,S)   (V transposed)
// MODE 2: f32 store (M,N) + residual  (output projection)
// ---------------------------------------------------------------------------
template <int MODE>
__global__ __launch_bounds__(256) void gemm_wmma(const _Float16* __restrict__ A,
                                                 const _Float16* __restrict__ Wt,
                                                 const float* __restrict__ bias,
                                                 const float* __restrict__ resid,
                                                 _Float16* __restrict__ outh,
                                                 float* __restrict__ outf) {
    constexpr int BM = 128, BN = 128, BK = 32, LD = BK + 8;  // LD=40 halves
    __shared__ __align__(16) _Float16 As[2][BM * LD];
    __shared__ __align__(16) _Float16 Bs[2][BN * LD];

    const int n0   = blockIdx.x * BN;
    const int m0   = blockIdx.y * BM;
    const int tid  = threadIdx.x;
    const int wave = tid >> 5;
    const int lane = tid & 31;
    const int wm   = wave >> 1;      // 0..3 : 32-row strip
    const int wn   = wave & 1;       // 0..1 : 64-col strip

    v8f acc0[4] = {}, acc1[4] = {};

    auto stage = [&](int kc, int buf) {
        // A: 128 rows x 32 halves = 512 x 16B ; B: same
#pragma unroll
        for (int i = 0; i < 2; ++i) {
            int idx = tid + i * 256, row = idx >> 2, slot = idx & 3;
            g2l_16B(&As[buf][row * LD + slot * 8],
                    &A[(size_t)(m0 + row) * HDIM + kc * BK + slot * 8]);
        }
#pragma unroll
        for (int i = 0; i < 2; ++i) {
            int idx = tid + i * 256, row = idx >> 2, slot = idx & 3;
            g2l_16B(&Bs[buf][row * LD + slot * 8],
                    &Wt[(size_t)(n0 + row) * HDIM + kc * BK + slot * 8]);
        }
    };

    stage(0, 0);
    wait_async();
    __syncthreads();

    for (int kc = 0; kc < HDIM / BK; ++kc) {
        const int cur = kc & 1;
        if (kc + 1 < HDIM / BK) stage(kc + 1, cur ^ 1);  // overlap DMA w/ math

        // preload ALL fragments, then 8 independent back-to-back WMMAs
        v16h a0 = frag_a(&As[cur][(wm * 32) * LD], LD);
        v16h a1 = frag_a(&As[cur][(wm * 32 + 16) * LD], LD);
        v16h bb[4];
#pragma unroll
        for (int nt = 0; nt < 4; ++nt)
            bb[nt] = frag_b(&Bs[cur][(wn * 64 + nt * 16) * LD], LD);
#pragma unroll
        for (int nt = 0; nt < 4; ++nt) acc0[nt] = wmma_f16(a0, bb[nt], acc0[nt]);
#pragma unroll
        for (int nt = 0; nt < 4; ++nt) acc1[nt] = wmma_f16(a1, bb[nt], acc1[nt]);

        wait_async();
        __syncthreads();
    }

    // epilogue. C layout: vgpr r, lanes0-15 -> row r, lanes16-31 -> row r+8
    const int lane15 = lane & 15;
#pragma unroll
    for (int mi = 0; mi < 2; ++mi) {
        const v8f* accp = mi ? acc1 : acc0;
        const int rbase = m0 + wm * 32 + mi * 16 + ((lane >> 4) << 3);
#pragma unroll
        for (int nt = 0; nt < 4; ++nt) {
            const int n  = n0 + wn * 64 + nt * 16 + lane15;
            const float bv = bias[n];
#pragma unroll
            for (int r = 0; r < 8; ++r) {
                const int m = rbase + r;
                float v = accp[nt][r] + bv;
                if (MODE == 0) {
                    int s = m >> 1, b = m & 1, nh = n >> 6, hd = n & 63;
                    outh[(((b * NHEAD + nh) * SDIM) + s) * HD + hd] = (_Float16)v;
                } else if (MODE == 1) {
                    int s = m >> 1, b = m & 1, nh = n >> 6, hd = n & 63;
                    outh[(((b * NHEAD + nh) * HD) + hd) * SDIM + s] = (_Float16)v;
                } else {
                    outf[(size_t)m * HDIM + n] = v + resid[(size_t)m * HDIM + n];
                }
            }
        }
    }
}

// ---------------------------------------------------------------------------
// Flash attention: block = (64 query rows, one head), 4 waves x 16-row strips.
// Q,K (B,NH,S,HD) f16 ; V transposed (B,NH,HD,S) f16 ; ctx -> f16 (M,H)
// K/V tiles double-buffered with async LDS copies; B fragments preloaded per
// K-chunk so 4 WMMAs issue per DS wait.
// ---------------------------------------------------------------------------
__global__ __launch_bounds__(128) void flash_attn(const _Float16* __restrict__ Q,
                                                  const _Float16* __restrict__ K,
                                                  const _Float16* __restrict__ Vt,
                                                  const float* __restrict__ mask,
                                                  _Float16* __restrict__ ctx) {
    constexpr int LDS_ = HD + 8;   // 72, row stride 144B (16B aligned)
    __shared__ __align__(16) _Float16 Qs[64 * LDS_];
    __shared__ __align__(16) _Float16 Ks[2][64 * LDS_];
    __shared__ __align__(16) _Float16 Vs[2][64 * LDS_];
    __shared__ __align__(16) _Float16 Ps[64 * LDS_];   // 4 waves x 16 rows

    const int tid = threadIdx.x, wave = tid >> 5, lane = tid & 31;
    const int q0  = blockIdx.x * 64;
    const int bh  = blockIdx.y;
    const int b   = bh / NHEAD, nh = bh % NHEAD;
    const float scale = 0.125f;   // 1/sqrt(64)

    const _Float16* Qh = Q  + (size_t)(b * NHEAD + nh) * SDIM * HD;
    const _Float16* Kh = K  + (size_t)(b * NHEAD + nh) * SDIM * HD;
    const _Float16* Vh = Vt + (size_t)(b * NHEAD + nh) * HD * SDIM;

    auto stage_kv = [&](int jb, int buf) {
        // 64 rows x 64 halves = 512 x 16B per matrix, 128 threads x 4
#pragma unroll
        for (int i = 0; i < 4; ++i) {
            int idx = tid + i * 128, row = idx >> 3, slot = idx & 7;
            g2l_16B(&Ks[buf][row * LDS_ + slot * 8],
                    &Kh[(size_t)(jb * 64 + row) * HD + slot * 8]);
        }
#pragma unroll
        for (int i = 0; i < 4; ++i) {
            int idx = tid + i * 128, row = idx >> 3, slot = idx & 7;
            g2l_16B(&Vs[buf][row * LDS_ + slot * 8],
                    &Vh[(size_t)row * SDIM + jb * 64 + slot * 8]);
        }
    };

    // stage Q block + first K/V tile
#pragma unroll
    for (int i = 0; i < 4; ++i) {
        int idx = tid + i * 128, row = idx >> 3, slot = idx & 7;
        g2l_16B(&Qs[row * LDS_ + slot * 8],
                &Qh[(size_t)(q0 + row) * HD + slot * 8]);
    }
    stage_kv(0, 0);
    wait_async();
    __syncthreads();

    const v16h aq0 = frag_a(&Qs[(wave * 16) * LDS_], LDS_);        // K 0..31
    const v16h aq1 = frag_a(&Qs[(wave * 16) * LDS_ + 32], LDS_);   // K 32..63

    float mrow[8], lrow[8];
#pragma unroll
    for (int r = 0; r < 8; ++r) { mrow[r] = -1e30f; lrow[r] = 0.f; }
    v8f o[4] = {};

    for (int jb = 0; jb < SDIM / 64; ++jb) {
        const int cur = jb & 1;
        if (jb + 1 < SDIM / 64) stage_kv(jb + 1, cur ^ 1);  // overlap DMA

        // scores: 16x64 = 4 tiles; preload 4 B frags per chunk -> 4 WMMAs/wait
        v8f sc[4] = {};
        v16h bb[4];
#pragma unroll
        for (int nt = 0; nt < 4; ++nt)
            bb[nt] = frag_b(&Ks[cur][(nt * 16) * LDS_], LDS_);
#pragma unroll
        for (int nt = 0; nt < 4; ++nt) sc[nt] = wmma_f16(aq0, bb[nt], sc[nt]);
#pragma unroll
        for (int nt = 0; nt < 4; ++nt)
            bb[nt] = frag_b(&Ks[cur][(nt * 16) * LDS_ + 32], LDS_);
#pragma unroll
        for (int nt = 0; nt < 4; ++nt) sc[nt] = wmma_f16(aq1, bb[nt], sc[nt]);

#pragma unroll
        for (int nt = 0; nt < 4; ++nt) {
            float mk = mask[b * SDIM + jb * 64 + nt * 16 + (lane & 15)];
#pragma unroll
            for (int r = 0; r < 8; ++r) sc[nt][r] = sc[nt][r] * scale + mk;
        }

        // online softmax
        float nm[8], alpha[8];
#pragma unroll
        for (int r = 0; r < 8; ++r) {
            float v = mrow[r];
#pragma unroll
            for (int nt = 0; nt < 4; ++nt) v = fmaxf(v, sc[nt][r]);
            nm[r] = halfmax(v);
            alpha[r] = __expf(mrow[r] - nm[r]);
            mrow[r] = nm[r];
        }
#pragma unroll
        for (int nt = 0; nt < 4; ++nt)
#pragma unroll
            for (int r = 0; r < 8; ++r) sc[nt][r] = __expf(sc[nt][r] - nm[r]);
#pragma unroll
        for (int r = 0; r < 8; ++r) {
            float s = sc[0][r] + sc[1][r] + sc[2][r] + sc[3][r];
            lrow[r] = lrow[r] * alpha[r] + halfsum(s);
#pragma unroll
            for (int nt = 0; nt < 4; ++nt) o[nt][r] *= alpha[r];
        }

        // P -> per-wave LDS area, then reload as A fragments
        _Float16* pw = &Ps[(wave * 16) * LDS_];
        const int prow = (lane >> 4) << 3, pcol = lane & 15;
#pragma unroll
        for (int nt = 0; nt < 4; ++nt)
#pragma unroll
            for (int r = 0; r < 8; ++r)
                pw[(prow + r) * LDS_ + nt * 16 + pcol] = (_Float16)sc[nt][r];
        asm volatile("s_wait_dscnt 0" ::: "memory");   // wave-local LDS RAW

        v16h pa0 = frag_a(pw, LDS_);
        v16h pa1 = frag_a(pw + 32, LDS_);
#pragma unroll
        for (int nt = 0; nt < 4; ++nt)
            bb[nt] = frag_b(&Vs[cur][(nt * 16) * LDS_], LDS_);
#pragma unroll
        for (int nt = 0; nt < 4; ++nt) o[nt] = wmma_f16(pa0, bb[nt], o[nt]);
#pragma unroll
        for (int nt = 0; nt < 4; ++nt)
            bb[nt] = frag_b(&Vs[cur][(nt * 16) * LDS_ + 32], LDS_);
#pragma unroll
        for (int nt = 0; nt < 4; ++nt) o[nt] = wmma_f16(pa1, bb[nt], o[nt]);

        wait_async();
        __syncthreads();   // protect K/V ping-pong buffers
    }

    // normalize + store ctx as (M, H) f16
    const int rbase = q0 + wave * 16 + ((lane >> 4) << 3);
#pragma unroll
    for (int r = 0; r < 8; ++r) {
        const float inv = 1.0f / lrow[r];
        const int s = rbase + r;
        const int t = s * BDIM + b;
#pragma unroll
        for (int nt = 0; nt < 4; ++nt) {
            int h = nh * HD + nt * 16 + (lane & 15);
            ctx[(size_t)t * HDIM + h] = (_Float16)(o[nt][r] * inv);
        }
    }
}

// ---------------------------------------------------------------------------
// LayerNorm over H=1024 per token; writes fp32 residual stream and f16 acts
// ---------------------------------------------------------------------------
__global__ __launch_bounds__(256) void layernorm(const float* __restrict__ in,
                                                 const float* __restrict__ g,
                                                 const float* __restrict__ be,
                                                 float* __restrict__ outf,
                                                 _Float16* __restrict__ outh) {
    __shared__ float red[256];
    const int t = blockIdx.x, tid = threadIdx.x;
    const float* row = in + (size_t)t * HDIM;
    float v[4], s = 0.f;
#pragma unroll
    for (int k = 0; k < 4; ++k) { v[k] = row[tid + k * 256]; s += v[k]; }
    red[tid] = s; __syncthreads();
    for (int off = 128; off > 0; off >>= 1) {
        if (tid < off) red[tid] += red[tid + off];
        __syncthreads();
    }
    const float mean = red[0] * (1.0f / HDIM);
    __syncthreads();
    float c[4]; s = 0.f;
#pragma unroll
    for (int k = 0; k < 4; ++k) { c[k] = v[k] - mean; s += c[k] * c[k]; }
    red[tid] = s; __syncthreads();
    for (int off = 128; off > 0; off >>= 1) {
        if (tid < off) red[tid] += red[tid + off];
        __syncthreads();
    }
    const float rs = rsqrtf(red[0] * (1.0f / HDIM) + 1e-12f);
#pragma unroll
    for (int k = 0; k < 4; ++k) {
        int i = tid + k * 256;
        float y = c[k] * rs * g[i] + be[i];
        outf[(size_t)t * HDIM + i] = y;
        if (outh) outh[(size_t)t * HDIM + i] = (_Float16)y;
    }
}

// ---------------------------------------------------------------------------
// host orchestration
// ---------------------------------------------------------------------------
extern "C" void kernel_launch(void* const* d_in, const int* in_sizes, int n_in,
                              void* d_out, int out_size, void* d_ws, size_t ws_size,
                              hipStream_t stream) {
    const float* x_in = (const float*)d_in[0];
    const float* mask = (const float*)d_in[1];
    const float* Wq = (const float*)d_in[2];
    const float* bq = (const float*)d_in[3];
    const float* Wk = (const float*)d_in[4];
    const float* bk = (const float*)d_in[5];
    const float* Wv = (const float*)d_in[6];
    const float* bv = (const float*)d_in[7];
    const float* Wo = (const float*)d_in[8];
    const float* bo = (const float*)d_in[9];
    const float* gamma = (const float*)d_in[10];
    const float* beta  = (const float*)d_in[11];
    float* out = (float*)d_out;

    const size_t MH  = (size_t)MTOK * HDIM;
    const size_t HH  = (size_t)HDIM * HDIM;
    char* ws = (char*)d_ws;
    _Float16* X16  = (_Float16*)(ws);                         // 8 MB
    _Float16* WtQ  = (_Float16*)(ws + MH * 2);                // 2 MB each
    _Float16* WtK  = WtQ + HH;
    _Float16* WtV  = WtK + HH;
    _Float16* WtO  = WtV + HH;
    _Float16* Q16  = WtO + HH;                                // 8 MB
    _Float16* K16  = Q16 + MH;
    _Float16* Vt16 = K16 + MH;
    _Float16* C16  = Vt16 + MH;
    float* dense   = (float*)(C16 + MH);                      // 16 MB
    float* xres    = dense + MH;                              // 16 MB

    // x -> f16
    cvt_f32_f16<<<(int)(MH / 256), 256, 0, stream>>>(x_in, X16, (int)MH);

    const dim3 gG(HDIM / 128, MTOK / 128);    // GEMM grid (8 x 32)
    const dim3 gW(HDIM / 32, HDIM / 32);      // weight transpose grid
    const dim3 gF(SDIM / 64, BDIM * NHEAD);   // flash grid

    for (int l = 0; l < NLAYER; ++l) {
        const float* res = (l == 0) ? x_in : xres;
        cvt_w_t<<<gW, 256, 0, stream>>>(Wq + l * HH, WtQ);
        cvt_w_t<<<gW, 256, 0, stream>>>(Wk + l * HH, WtK);
        cvt_w_t<<<gW, 256, 0, stream>>>(Wv + l * HH, WtV);
        cvt_w_t<<<gW, 256, 0, stream>>>(Wo + l * HH, WtO);

        gemm_wmma<0><<<gG, 256, 0, stream>>>(X16, WtQ, bq + l * HDIM, nullptr, Q16, nullptr);
        gemm_wmma<0><<<gG, 256, 0, stream>>>(X16, WtK, bk + l * HDIM, nullptr, K16, nullptr);
        gemm_wmma<1><<<gG, 256, 0, stream>>>(X16, WtV, bv + l * HDIM, nullptr, Vt16, nullptr);

        flash_attn<<<gF, 128, 0, stream>>>(Q16, K16, Vt16, mask, C16);

        gemm_wmma<2><<<gG, 256, 0, stream>>>(C16, WtO, bo + l * HDIM, res, nullptr, dense);

        const bool last = (l == NLAYER - 1);
        layernorm<<<MTOK, 256, 0, stream>>>(dense, gamma + l * HDIM, beta + l * HDIM,
                                            last ? out : xres, last ? nullptr : X16);
    }
}
